// TinyLSTM_36979668418797
// MI455X (gfx1250) — matrix operand order repaired and checked by reading, not compiled, not run on hardware
//
#include <hip/hip_runtime.h>
#include <hip/hip_bf16.h>

// ---------------------------------------------------------------------------
// TinyLSTM for MI455X (gfx1250): bf16 WMMA, wave32, persistent recurrence.
// Phases: transpose/convert weights -> GEMM0(x@Wx0) -> persistent layer0
//         -> GEMM1(h0@Wx1) -> persistent layer1 -> GEMM2(h1@Why) -> fin.
// ---------------------------------------------------------------------------

#define BB 32
#define TT 512
#define HH 1024
#define OO 1024
#define LL 2
#define G4 4096   // 4*HH

#define REC_BLOCKS 16   // persistent recurrence: 16 blocks x 4 waves = 64 waves

typedef __attribute__((ext_vector_type(16))) __bf16       v16bf;
typedef __attribute__((ext_vector_type(8)))  float        v8f;
typedef __attribute__((ext_vector_type(4)))  unsigned int u32x4;

union Frag128x2 { u32x4 q[2]; v16bf v; };

static __device__ inline v8f vzero8() {
  v8f z;
#pragma unroll
  for (int i = 0; i < 8; ++i) z[i] = 0.0f;
  return z;
}

// A fragment (16x32 bf16, row-major A[M,K]): lane holds row m0+(lane&15),
// VGPR0-3 = k+[(lane>>4)*8 .. +7], VGPR4-7 = +16. Two b128 loads.
static __device__ inline v16bf load_a_frag(const __bf16* __restrict__ A, int lda,
                                           int m0, int kk, int lane) {
  const int m    = m0 + (lane & 15);
  const int koff = (lane >> 4) * 8;
  const __bf16* p = A + (size_t)m * lda + kk + koff;
  Frag128x2 f;
  f.q[0] = *(const u32x4*)(p);
  f.q[1] = *(const u32x4*)(p + 16);
  return f.v;
}

// B fragment (32x16 bf16) from pre-transposed BT[N,K] row-major: lane holds
// column n0+(lane&15), VGPR0-3 = k+[(lane>>4)*8 .. +7], VGPR4-7 = +16. Two b128 loads.
static __device__ inline v16bf load_b_frag(const __bf16* __restrict__ BT, int ldk,
                                           int n0, int kk, int lane) {
  const int n    = n0 + (lane & 15);
  const int koff = (lane >> 4) * 8;
  const __bf16* p = BT + (size_t)n * ldk + kk + koff;
  Frag128x2 f;
  f.q[0] = *(const u32x4*)(p);
  f.q[1] = *(const u32x4*)(p + 16);
  return f.v;
}

static __device__ inline float sigmoidf_fast(float x) {
  return 1.0f / (1.0f + __expf(-x));
}

// Device-scope grid barrier for the persistent recurrence (REC_BLOCKS blocks,
// all co-resident). Monotonic counter: target = REC_BLOCKS * (step+1).
static __device__ inline void grid_sync(unsigned* cnt, unsigned target) {
  __threadfence();
  __syncthreads();
  if (threadIdx.x == 0) {
    __hip_atomic_fetch_add(cnt, 1u, __ATOMIC_ACQ_REL, __HIP_MEMORY_SCOPE_AGENT);
    while (__hip_atomic_load(cnt, __ATOMIC_ACQUIRE, __HIP_MEMORY_SCOPE_AGENT) <
           target) {
      __builtin_amdgcn_s_sleep(2);
    }
  }
  __syncthreads();
  __threadfence();
}

// ---------------------------------------------------------------------------
// Generic bf16 WMMA GEMM: out = A[M,K] * BT[N,K]^T + bias.
// Block = 256 threads = 8 waves (2 M x 4 N), wave tile 32x64, block tile 64x256.
// mode 0: write bf16 row-major [M,N] (gate pre-activations Xg)
// mode 1: write f32 with row m=(t*32+b) scattered to out[b][t][n] (y proj)
// ---------------------------------------------------------------------------
__global__ void __launch_bounds__(256)
lstm_gemm_bf16_wmma(const __bf16* __restrict__ A, const __bf16* __restrict__ BT,
                    const float* __restrict__ bias, __bf16* __restrict__ outb,
                    float* __restrict__ outf, int M, int N, int K,
                    int mode, int Tdim) {
  const int lane  = threadIdx.x & 31;
  const int wave  = threadIdx.x >> 5;
  const int waveM = wave >> 2;
  const int waveN = wave & 3;
  const int bm = blockIdx.x * 64 + waveM * 32;
  const int bn = blockIdx.y * 256 + waveN * 64;

  v8f acc[2][4];
#pragma unroll
  for (int mf = 0; mf < 2; ++mf)
#pragma unroll
    for (int nf = 0; nf < 4; ++nf) acc[mf][nf] = vzero8();

  for (int kk = 0; kk < K; kk += 32) {
    if (kk + 32 < K) {  // speculative prefetch of next K-slab (global_prefetch_b8)
      __builtin_prefetch(A + (size_t)(bm + (lane & 15)) * K + kk + 32, 0, 1);
      __builtin_prefetch(BT + (size_t)(bn + (lane & 15)) * K + kk + 32, 0, 1);
    }
    v16bf a[2];
    a[0] = load_a_frag(A, K, bm + 0, kk, lane);
    a[1] = load_a_frag(A, K, bm + 16, kk, lane);
    v16bf bf[4];
#pragma unroll
    for (int nf = 0; nf < 4; ++nf) bf[nf] = load_b_frag(BT, K, bn + nf * 16, kk, lane);
#pragma unroll
    for (int mf = 0; mf < 2; ++mf)
#pragma unroll
      for (int nf = 0; nf < 4; ++nf)
        acc[mf][nf] = __builtin_amdgcn_wmma_f32_16x16x32_bf16(
            false, a[mf], false, bf[nf], (short)0, acc[mf][nf], false, false);
  }

  const int half = lane >> 4;
  const int nc   = lane & 15;
  if (mode == 0) {
#pragma unroll
    for (int nf = 0; nf < 4; ++nf)
#pragma unroll
      for (int mf = 0; mf < 2; ++mf)
#pragma unroll
        for (int r = 0; r < 8; ++r) {
          const int m = bm + mf * 16 + half * 8 + r;
          const int n = bn + nf * 16 + nc;
          outb[(size_t)m * N + n] = (__bf16)(acc[mf][nf][r] + bias[n]);
        }
  } else {
#pragma unroll
    for (int nf = 0; nf < 4; ++nf)
#pragma unroll
      for (int mf = 0; mf < 2; ++mf)
#pragma unroll
        for (int r = 0; r < 8; ++r) {
          const int m = bm + mf * 16 + half * 8 + r;
          const int n = bn + nf * 16 + nc;
          const int t = m >> 5, b = m & 31;  // Bdim == 32 always
          outf[(size_t)b * Tdim * N + (size_t)t * N + n] = acc[mf][nf][r] + bias[n];
        }
  }
}

// ---------------------------------------------------------------------------
// Persistent recurrence for one whole layer: loops over all TT timesteps with
// a device-scope grid barrier between steps. 16 blocks x 128 threads = 64
// waves; wave gw owns feature cols j0=gw*16 and computes all 4 gate tiles
// (N = g*1024 + j0) of h_prev[32,1024] @ W_hT per step. Cell state for the
// wave's (32 rows x 16 cols) patch lives in 16 VGPRs per lane for all steps;
// c_seq is write-only. h goes through a bf16 ring in ws (slot t -> slot t+1).
// ---------------------------------------------------------------------------
__global__ void __launch_bounds__(128)
lstm_layer_persistent(const __bf16* __restrict__ hbf,  // ring [(T+1)][32][1024]
                      const __bf16* __restrict__ whT,  // [4096,1024]
                      const __bf16* __restrict__ Xg,   // [T*32,4096] t-major
                      float* __restrict__ c_l,         // c_seq layer base
                      float* __restrict__ h_l,         // h_seq layer base
                      __bf16* __restrict__ hbf_w,      // same ring, writable
                      unsigned* __restrict__ bar) {
  const int lane = threadIdx.x & 31;
  const int gw   = blockIdx.x * 4 + (threadIdx.x >> 5);
  const int j0   = gw * 16;
  const int half = lane >> 4;
  const int j    = j0 + (lane & 15);

  float cst[2][8];
#pragma unroll
  for (int mf = 0; mf < 2; ++mf)
#pragma unroll
    for (int r = 0; r < 8; ++r) cst[mf][r] = 0.0f;

  for (int t = 0; t < TT; ++t) {
    const __bf16* hprev = hbf + (size_t)t * BB * HH;

    v8f acc[4][2];
#pragma unroll
    for (int g = 0; g < 4; ++g) { acc[g][0] = vzero8(); acc[g][1] = vzero8(); }

    for (int kk = 0; kk < HH; kk += 32) {
      v16bf a0 = load_a_frag(hprev, HH, 0, kk, lane);
      v16bf a1 = load_a_frag(hprev, HH, 16, kk, lane);
#pragma unroll
      for (int g = 0; g < 4; ++g) {
        v16bf bg = load_b_frag(whT, HH, g * 1024 + j0, kk, lane);
        acc[g][0] = __builtin_amdgcn_wmma_f32_16x16x32_bf16(
            false, a0, false, bg, (short)0, acc[g][0], false, false);
        acc[g][1] = __builtin_amdgcn_wmma_f32_16x16x32_bf16(
            false, a1, false, bg, (short)0, acc[g][1], false, false);
      }
    }

    float* cout = c_l + (size_t)t * BB * HH;
    float* hout = h_l + (size_t)t * BB * HH;
    __bf16* hnx = hbf_w + (size_t)(t + 1) * BB * HH;
#pragma unroll
    for (int mf = 0; mf < 2; ++mf)
#pragma unroll
      for (int r = 0; r < 8; ++r) {
        const int brow = mf * 16 + half * 8 + r;  // batch index 0..31
        const __bf16* xr = Xg + (size_t)(t * BB + brow) * G4;
        const float iv = sigmoidf_fast(acc[0][mf][r] + (float)xr[j]);
        const float fv = sigmoidf_fast(acc[1][mf][r] + (float)xr[1024 + j]);
        const float ov = sigmoidf_fast(acc[2][mf][r] + (float)xr[2048 + j]);
        const float gv = tanhf(acc[3][mf][r] + (float)xr[3072 + j]);
        const float cn = fv * cst[mf][r] + iv * gv;
        const float hn = ov * tanhf(cn);
        cst[mf][r] = cn;
        cout[brow * HH + j] = cn;
        hout[brow * HH + j] = hn;
        hnx[brow * HH + j]  = (__bf16)hn;
      }

    if (t + 1 < TT) grid_sync(bar, (unsigned)(REC_BLOCKS * (t + 1)));
  }
}

// ---------------------------------------------------------------------------
// Helper kernels: transpose+convert weights, permute x, zero init, finalize.
// ---------------------------------------------------------------------------
__global__ void transpose_cvt_bf16(const float* __restrict__ in,  // [R,C]
                                   __bf16* __restrict__ out,      // [C,R]
                                   int R, int C) {
  __shared__ float tile[32][33];
  const int c0 = blockIdx.x * 32, r0 = blockIdx.y * 32;
  const int tx = threadIdx.x, ty = threadIdx.y;  // 32 x 8
#pragma unroll
  for (int i = 0; i < 32; i += 8)
    tile[ty + i][tx] = in[(size_t)(r0 + ty + i) * C + c0 + tx];
  __syncthreads();
#pragma unroll
  for (int i = 0; i < 32; i += 8)
    out[(size_t)(c0 + ty + i) * R + r0 + tx] = (__bf16)tile[tx][ty + i];
}

__global__ void permute_x_bf16(const float* __restrict__ x,  // [B,T,H]
                               __bf16* __restrict__ xbf) {   // [T,B,H]
  const size_t idx = (size_t)blockIdx.x * blockDim.x + threadIdx.x;
  if (idx >= (size_t)BB * TT * HH) return;
  const int h = (int)(idx % HH);
  const int row = (int)(idx / HH);
  const int b = row / TT, t = row % TT;
  xbf[((size_t)t * BB + b) * HH + h] = (__bf16)x[idx];
}

__global__ void init_state(__bf16* __restrict__ h0, int n,
                           unsigned* __restrict__ bars) {
  const int i = blockIdx.x * blockDim.x + threadIdx.x;
  if (i < n) h0[i] = (__bf16)0.0f;
  if (i < LL) bars[i] = 0u;
}

__global__ void finalize_hc(const float* __restrict__ h_seq,  // [L,T,B,H]
                            const float* __restrict__ c_seq,
                            float* __restrict__ h_fin,        // [L,B,H]
                            float* __restrict__ c_fin) {
  const int idx = blockIdx.x * blockDim.x + threadIdx.x;
  if (idx >= LL * BB * HH) return;
  const int l = idx / (BB * HH);
  const int rem = idx % (BB * HH);
  const size_t src = ((size_t)l * TT + (TT - 1)) * BB * HH + rem;
  h_fin[idx] = h_seq[src];
  c_fin[idx] = c_seq[src];
}

// ---------------------------------------------------------------------------
extern "C" void kernel_launch(void* const* d_in, const int* in_sizes, int n_in,
                              void* d_out, int out_size, void* d_ws, size_t ws_size,
                              hipStream_t stream) {
  (void)in_sizes; (void)n_in; (void)out_size; (void)ws_size;
  const float* x    = (const float*)d_in[0];  // [B,T,H]
  const float* W_x  = (const float*)d_in[1];  // [L,H,4H]
  const float* W_h  = (const float*)d_in[2];  // [L,H,4H]
  const float* bias = (const float*)d_in[3];  // [L,4H]
  const float* W_hy = (const float*)d_in[4];  // [H,O]
  const float* b_y  = (const float*)d_in[5];  // [O]

  // d_out partition (fp32): y [B,T,O] | h_seq [L,T,B,H] | c_seq | h_fin | c_fin
  float* out_y = (float*)d_out;
  float* h_seq = out_y + (size_t)BB * TT * OO;
  float* c_seq = h_seq + (size_t)LL * TT * BB * HH;
  float* h_fin = c_seq + (size_t)LL * TT * BB * HH;
  float* c_fin = h_fin + (size_t)LL * BB * HH;

  // workspace partition
  __bf16* wxT  = (__bf16*)d_ws;                       // [L][4096][1024]
  __bf16* whT  = wxT + (size_t)LL * G4 * HH;          // [L][4096][1024]
  __bf16* whyT = whT + (size_t)LL * G4 * HH;          // [1024][1024]
  __bf16* xbf  = whyT + (size_t)OO * HH;              // [T*B][1024]
  __bf16* hbf  = xbf + (size_t)TT * BB * HH;          // [(T+1)][B][H] ring
  __bf16* Xg   = hbf + (size_t)(TT + 1) * BB * HH;    // [T*B][4096]
  unsigned* bars = (unsigned*)(Xg + (size_t)TT * BB * G4);  // [L] barrier cnts

  // --- phase 0: convert / transpose weights & inputs -----------------------
  dim3 tb(32, 8);
  for (int l = 0; l < LL; ++l) {
    transpose_cvt_bf16<<<dim3(G4 / 32, HH / 32), tb, 0, stream>>>(
        W_x + (size_t)l * HH * G4, wxT + (size_t)l * G4 * HH, HH, G4);
    transpose_cvt_bf16<<<dim3(G4 / 32, HH / 32), tb, 0, stream>>>(
        W_h + (size_t)l * HH * G4, whT + (size_t)l * G4 * HH, HH, G4);
  }
  transpose_cvt_bf16<<<dim3(OO / 32, HH / 32), tb, 0, stream>>>(W_hy, whyT, HH, OO);
  {
    const size_t n = (size_t)BB * TT * HH;
    permute_x_bf16<<<(unsigned)((n + 255) / 256), 256, 0, stream>>>(x, xbf);
  }
  init_state<<<(BB * HH + 255) / 256, 256, 0, stream>>>(hbf, BB * HH, bars);

  const int Mrows = TT * BB;  // 16384
  const dim3 gemm_block(256);

  for (int l = 0; l < LL; ++l) {
    // --- input-gate GEMM for this layer: Xg = inp_seq @ W_x[l] + b[l] ------
    const __bf16* Aseq = (l == 0) ? xbf : (hbf + (size_t)BB * HH);  // slots 1..T
    lstm_gemm_bf16_wmma<<<dim3(Mrows / 64, G4 / 256), gemm_block, 0, stream>>>(
        Aseq, wxT + (size_t)l * G4 * HH, bias + (size_t)l * G4,
        Xg, nullptr, Mrows, G4, HH, /*mode=*/0, TT);

    // --- persistent recurrence over all timesteps --------------------------
    lstm_layer_persistent<<<REC_BLOCKS, 128, 0, stream>>>(
        hbf, whT + (size_t)l * G4 * HH, Xg,
        c_seq + (size_t)l * TT * BB * HH, h_seq + (size_t)l * TT * BB * HH,
        hbf, bars + l);
  }

  // --- output projection: y = h_seq(L-1) @ W_hy + b_y (scatter t,b -> b,t) --
  lstm_gemm_bf16_wmma<<<dim3(Mrows / 64, OO / 256), gemm_block, 0, stream>>>(
      hbf + (size_t)BB * HH, whyT, b_y, nullptr, out_y,
      Mrows, OO, HH, /*mode=*/1, TT);

  finalize_hc<<<(LL * BB * HH + 255) / 256, 256, 0, stream>>>(h_seq, c_seq,
                                                              h_fin, c_fin);
}
